// gnn_model_87582973100471
// MI455X (gfx1250) — matrix operand order repaired and checked by reading, not compiled, not run on hardware
//
#include <hip/hip_runtime.h>
#include <hip/hip_bf16.h>

// GNN (spektral GeneralGNN-style) forward on MI455X (gfx1250, wave32).
// Every Dense->BN->sigmoid is one v_wmma_f32_16x16x32_f16 tile per wave
// (all output widths are 16). BN+bias folded into pre-packed f16 B-fragments.
// Messages z stored f16 (halves gather bytes, cvt-free A loads); aggregation
// and pooling via f32 global atomics (L2-resident working set).
// Sigmoid uses v_exp_f32 + v_rcp_f32 (no exact-division sequence).

typedef __attribute__((ext_vector_type(16))) _Float16 v16h;
typedef __attribute__((ext_vector_type(8)))  _Float16 v8h;
typedef __attribute__((ext_vector_type(2)))  _Float16 v2h;
typedef __attribute__((ext_vector_type(8)))  float    v8f;

#define GNN_N   262144
#define GNN_E   4194304
#define GNN_B   512
#define BN_EPS  1e-3f

static __device__ __forceinline__ float sigmoidf(float t) {
    // v_exp_f32 then v_rcp_f32 -- avoids the IEEE div_scale/div_fmas chain.
    return __builtin_amdgcn_rcpf(1.0f + __expf(-t));
}

// ---------------------------------------------------------------- zero ws --
__global__ void k_zero(float4* __restrict__ p, size_t n4) {
    size_t i      = (size_t)blockIdx.x * blockDim.x + threadIdx.x;
    size_t stride = (size_t)gridDim.x * blockDim.x;
    float4 z; z.x = z.y = z.z = z.w = 0.0f;
    for (; i < n4; i += stride) p[i] = z;
}

// ----------------------------------------------- weight pre-pack (1 wave) --
// Weff[k][n] = W[k][n] * gamma[n]/sqrt(var[n]+eps) packed in the CDNA5 WMMA
// f16 B-fragment layout (per 32-K block: lane holds col N=lane&15, element j
// holds K = 8h+j (j<8) | 16+8h+(j-8) (j>=8), h = half-wave). ceff[n] folds bias+BN shift.
__global__ __launch_bounds__(32)
void k_prepack(const float* __restrict__ W, const float* __restrict__ bias,
               const float* __restrict__ gamma, const float* __restrict__ beta,
               const float* __restrict__ mean, const float* __restrict__ var,
               int Kin, int nkb, _Float16* __restrict__ packB,
               float* __restrict__ ceff) {
    int   lane = threadIdx.x;
    int   col  = lane & 15;
    float s    = gamma[col] * rsqrtf(var[col] + BN_EPS);
    if (lane < 16)
        ceff[lane] = (bias[lane] - mean[lane]) * s + beta[lane];
    for (int kb = 0; kb < nkb; ++kb) {
        _Float16* dst = packB + ((size_t)kb * 32 + lane) * 16;
        for (int j = 0; j < 16; ++j) {
            int   K = kb * 32 + ((j < 8) ? 8 * (lane >> 4) + j : 16 + 8 * (lane >> 4) + (j - 8));
            float v = (K < Kin) ? W[(size_t)K * 16 + col] * s : 0.0f;
            dst[j] = (_Float16)v;
        }
    }
}

// A-fragment loader (16x32 f16 tile, lane holds row M=lane&15).
// K = kb*32 + 2p + (p<4 ? 0 : 8) + hi*8  (consecutive-K pairs).
// Pad predicate is per-16-chunk and identical for both half-waves (every KIN
// is a multiple of 16), so with compile-time KIN/kb it folds to straight-line
// clause-able loads — no EXEC divergence, no per-load waits.
template <int KIN, bool IN16>
static __device__ __forceinline__ v16h load_afrag(const void* __restrict__ rowv,
                                                  int kb, int hi) {
    const float*    rowf = (const float*)rowv;
    const _Float16* rowh = (const _Float16*)rowv;
    v16h a;
#pragma unroll
    for (int p = 0; p < 8; ++p) {
        const int chunk = kb * 2 + ((p < 4) ? 0 : 1);
        if (chunk * 16 < KIN) {
            int K = kb * 32 + 2 * p + ((p < 4) ? 0 : 8) + hi * 8;
            if (IN16) {
                v2h v = *(const v2h*)(rowh + K);
                a[2 * p]     = v.x;
                a[2 * p + 1] = v.y;
            } else {
                float2 v = *(const float2*)(rowf + K);
                a[2 * p]     = (_Float16)v.x;
                a[2 * p + 1] = (_Float16)v.y;
            }
        } else {
            a[2 * p]     = (_Float16)0.0f;
            a[2 * p + 1] = (_Float16)0.0f;
        }
    }
    return a;
}

// --------------------------------------- Dense(K->16) + BN + sigmoid, WMMA --
// One wave per 16-node tile; 4 waves (128 threads) per block. Fully
// specialized per layer: unrolled K blocks, no divergence around the WMMA.
template <int KIN, int NKB, bool IN16, bool OUT16>
__global__ __launch_bounds__(128)
void k_dense16(const void* __restrict__ hinv, int istride,
               const _Float16* __restrict__ packB,
               const float* __restrict__ ceff,
               void* __restrict__ outv, int ostride) {
    int lane = threadIdx.x & 31;
    int wave = threadIdx.x >> 5;
    int tile = blockIdx.x * 4 + wave;
    int m    = lane & 15;
    int hi   = lane >> 4;

    const void* row;
    if (IN16) row = (const void*)((const _Float16*)hinv + (size_t)(tile * 16 + m) * istride);
    else      row = (const void*)((const float*)hinv    + (size_t)(tile * 16 + m) * istride);

    v8f acc = {};
#pragma unroll
    for (int kb = 0; kb < NKB; ++kb) {
        v16h a = load_afrag<KIN, IN16>(row, kb, hi);
        v16h b = *(const v16h*)(packB + ((size_t)kb * 32 + lane) * 16);
        acc = __builtin_amdgcn_wmma_f32_16x16x32_f16(
                  false, a, false, b, (short)0, acc, false, false);
    }

    float cv = ceff[m];
    if (OUT16) {
        _Float16* orow = (_Float16*)outv + (size_t)(tile * 16 + 8 * hi) * ostride + m;
#pragma unroll
        for (int r = 0; r < 8; ++r)
            orow[(size_t)r * ostride] = (_Float16)sigmoidf(acc[r] + cv);
    } else {
        float* orow = (float*)outv + (size_t)(tile * 16 + 8 * hi) * ostride + m;
#pragma unroll
        for (int r = 0; r < 8; ++r)
            orow[(size_t)r * ostride] = sigmoidf(acc[r] + cv);
    }
}

// ----------------------------------------- edge gather + scatter-sum ------
// thread = (edge, 8-col half-row): 16B f16 gather from z[src], f32 atomic
// accumulation into the agg columns of H (keeps sum precision in f32).
__global__ __launch_bounds__(256)
void k_edge_scatter(const int* __restrict__ src, const int* __restrict__ tgt,
                    const _Float16* __restrict__ Z, float* __restrict__ Hagg,
                    int total) {
    int idx = blockIdx.x * blockDim.x + threadIdx.x;
    if (idx >= total) return;
    int e = idx >> 1;
    int c = (idx & 1) << 3;
    int s = src[e];
    int t = tgt[e];
    v8h v = *(const v8h*)(Z + (size_t)s * 16 + c);
    float* d = Hagg + (size_t)t * 80 + c;
#pragma unroll
    for (int i = 0; i < 8; ++i)
        atomicAdd(d + i, (float)v[i]);
}

// -------------------------------------------------- global sum pooling ----
__global__ __launch_bounds__(256)
void k_pool(const float* __restrict__ H, const int* __restrict__ gidx,
            float* __restrict__ G, int total) {
    int idx = blockIdx.x * blockDim.x + threadIdx.x;
    if (idx >= total) return;
    int n = idx / 20;
    int c = (idx % 20) * 4;
    int g = gidx[n];
    float4 v = *(const float4*)(H + (size_t)n * 80 + c);
    float* d = G + (size_t)g * 80 + c;
    atomicAdd(d + 0, v.x);
    atomicAdd(d + 1, v.y);
    atomicAdd(d + 2, v.z);
    atomicAdd(d + 3, v.w);
}

// ------------------- post MLP: Dense(80->16)+BN+sig, Dense(16->1)+BN+sig ---
// One wave per 16-graph tile (32 tiles). Final [16->1] = 16-lane shfl dot.
__global__ __launch_bounds__(32)
void k_post(const float* __restrict__ G,
            const _Float16* __restrict__ packB, const float* __restrict__ ceff,
            const float* __restrict__ W1, const float* __restrict__ b1,
            const float* __restrict__ g1, const float* __restrict__ be1,
            const float* __restrict__ m1, const float* __restrict__ v1,
            float* __restrict__ out) {
    int lane = threadIdx.x;
    int tile = blockIdx.x;
    int m    = lane & 15;
    int hi   = lane >> 4;

    const float* row = G + (size_t)(tile * 16 + m) * 80;
    v8f acc = {};
#pragma unroll
    for (int kb = 0; kb < 3; ++kb) {           // K=80 padded to 96
        v16h a = load_afrag<80, false>(row, kb, hi);
        v16h b = *(const v16h*)(packB + ((size_t)kb * 32 + lane) * 16);
        acc = __builtin_amdgcn_wmma_f32_16x16x32_f16(
                  false, a, false, b, (short)0, acc, false, false);
    }
    float cv  = ceff[m];
    float s1  = g1[0] * rsqrtf(v1[0] + BN_EPS);
    float c1  = (b1[0] - m1[0]) * s1 + be1[0];
    float w1e = W1[m] * s1;                    // W1 shape [16,1]
#pragma unroll
    for (int r = 0; r < 8; ++r) {
        float y   = sigmoidf(acc[r] + cv);
        float val = y * w1e;
#pragma unroll
        for (int sh = 1; sh < 16; sh <<= 1)    // reduce 16 cols per half-wave
            val += __shfl_xor(val, sh, 32);
        if (m == 0)
            out[tile * 16 + 8 * hi + r] = sigmoidf(val + c1);
    }
}

// --------------------------------------------------------------- launcher --
extern "C" void kernel_launch(void* const* d_in, const int* in_sizes, int n_in,
                              void* d_out, int out_size, void* d_ws, size_t ws_size,
                              hipStream_t stream) {
    (void)in_sizes; (void)n_in; (void)out_size; (void)ws_size;

    const int N = GNN_N, E = GNN_E, B = GNN_B;

    // Input order (setup_inputs dict insertion order, params tree in order):
    // 0: x ; then 8 layers x {W,b,gamma,beta,mean,var} in order
    // pre0,pre1,conv0..3,post0,post1 ; then edge_src, edge_tgt, graph_idx, n.
    const float* x = (const float*)d_in[0];
    const float* P[8][6];
    for (int L = 0; L < 8; ++L)
        for (int k = 0; k < 6; ++k)
            P[L][k] = (const float*)d_in[1 + 6 * L + k];
    const int* edge_src  = (const int*)d_in[49];
    const int* edge_tgt  = (const int*)d_in[50];
    const int* graph_idx = (const int*)d_in[51];

    // Workspace layout:
    //   H [N,80] f32 : concat layout [agg4|agg3|agg2|agg1|h_pre], cols 0..79
    //   G [B,80] f32 : pooled graph features (contiguous after H for zeroing)
    //   Z [N,16] f16 : per-layer dense output / message buffer
    //   packB        : 11 kblocks * 512 f16 (pre0,pre1,conv0..3,post0)
    //   ceff         : 7 * 16 f32
    float*    H     = (float*)d_ws;
    float*    G     = H + (size_t)N * 80;
    _Float16* Z     = (_Float16*)(G + (size_t)B * 80);
    _Float16* packB = Z + (size_t)N * 16;
    float*    ceff  = (float*)(packB + 11 * 512);

    // zero H and G (contiguous)
    size_t n4 = ((size_t)(N + B) * 80) / 4;
    k_zero<<<4096, 256, 0, stream>>>((float4*)H, n4);

    // pre-pack effective weights for the 7 width-16 dense layers
    static const int kKin[7]   = {32, 16, 16, 32, 48, 64, 80};
    static const int kNkb[7]   = { 1,  1,  1,  1,  2,  2,  3};
    static const int kKbOff[7] = { 0,  1,  2,  3,  4,  6,  8};
    for (int L = 0; L < 7; ++L) {
        k_prepack<<<1, 32, 0, stream>>>(P[L][0], P[L][1], P[L][2], P[L][3],
                                        P[L][4], P[L][5], kKin[L], kNkb[L],
                                        packB + (size_t)kKbOff[L] * 512,
                                        ceff + (size_t)L * 16);
    }

    const int denseGrid = N / 16 / 4;   // 4 tiles (waves) per block

    // pre MLP: x[N,32] f32 -> Z f16 -> H cols 64..79 f32
    k_dense16<32, 1, false, true><<<denseGrid, 128, 0, stream>>>(
        x, 32, packB + (size_t)kKbOff[0] * 512, ceff + 0 * 16, Z, 16);
    k_dense16<16, 1, true, false><<<denseGrid, 128, 0, stream>>>(
        Z, 16, packB + (size_t)kKbOff[1] * 512, ceff + 1 * 16, H + 64, 80);

    // 4 GeneralConv layers with 'cat' skip: input cols [64-16l .. 79],
    // aggregate scattered into cols [48-16l .. 63-16l].
    const int edgeTotal = E * 2;        // (edge, 8-col half-row)
    const int edgeGrid  = (edgeTotal + 255) / 256;

    k_dense16<16, 1, false, true><<<denseGrid, 128, 0, stream>>>(
        H + 64, 80, packB + (size_t)kKbOff[2] * 512, ceff + 2 * 16, Z, 16);
    k_edge_scatter<<<edgeGrid, 256, 0, stream>>>(edge_src, edge_tgt, Z, H + 48, edgeTotal);

    k_dense16<32, 1, false, true><<<denseGrid, 128, 0, stream>>>(
        H + 48, 80, packB + (size_t)kKbOff[3] * 512, ceff + 3 * 16, Z, 16);
    k_edge_scatter<<<edgeGrid, 256, 0, stream>>>(edge_src, edge_tgt, Z, H + 32, edgeTotal);

    k_dense16<48, 2, false, true><<<denseGrid, 128, 0, stream>>>(
        H + 32, 80, packB + (size_t)kKbOff[4] * 512, ceff + 4 * 16, Z, 16);
    k_edge_scatter<<<edgeGrid, 256, 0, stream>>>(edge_src, edge_tgt, Z, H + 16, edgeTotal);

    k_dense16<64, 2, false, true><<<denseGrid, 128, 0, stream>>>(
        H + 16, 80, packB + (size_t)kKbOff[5] * 512, ceff + 5 * 16, Z, 16);
    k_edge_scatter<<<edgeGrid, 256, 0, stream>>>(edge_src, edge_tgt, Z, H + 0, edgeTotal);

    // global sum pool H[N,80] -> G[B,80]
    const int poolTotal = N * 20;
    k_pool<<<(poolTotal + 255) / 256, 256, 0, stream>>>(H, graph_idx, G, poolTotal);

    // post MLP -> out[B,1]
    k_post<<<B / 16, 32, 0, stream>>>(G,
                                      packB + (size_t)kKbOff[6] * 512,
                                      ceff + 6 * 16,
                                      P[7][0], P[7][1], P[7][2], P[7][3],
                                      P[7][4], P[7][5], (float*)d_out);
}
